// GTDynamics_2714419331263
// MI455X (gfx1250) — matrix operand order repaired and checked by reading, not compiled, not run on hardware
//
#include <hip/hip_runtime.h>
#include <hip/hip_bf16.h>
#include <math.h>

#define B_      2
#define N_      512
#define INNF_   3
#define DIM_    256
#define HEADS_  8
#define DHEAD_  32
#define INNER_  256
#define DEPTH_  2
#define SCALE_  0.17677669529663687f   /* 32^-0.5 */

typedef __attribute__((ext_vector_type(16))) _Float16 v16h;
typedef __attribute__((ext_vector_type(8)))  float    v8f;

__device__ __forceinline__ v8f wmma16(v16h a, v16h b, v8f c) {
  // D = A(16x32 f16) * B(32x16 f16) + C(16x16 f32)
  return __builtin_amdgcn_wmma_f32_16x16x32_f16(
      false, a, false, b, (short)0, c, false, false);
}

// ---------------------------------------------------------------- dist
__global__ void k_dist(const float* __restrict__ x, float* __restrict__ dist) {
  int idx = blockIdx.x * blockDim.x + threadIdx.x;
  if (idx >= B_ * N_ * N_) return;
  int j = idx % N_;
  int i = (idx / N_) % N_;
  int b = idx / (N_ * N_);
  const float* xi = x + (b * N_ + i) * INNF_;
  const float* xj = x + (b * N_ + j) * INNF_;
  float d0 = xi[0] - xj[0], d1 = xi[1] - xj[1], d2 = xi[2] - xj[2];
  float s = d0 * d0 + d1 * d1 + d2 * d2;
  dist[idx] = (s > 0.0f) ? sqrtf(s) : 0.0f;
}

// ---------------------------------------------------------------- encode
__global__ void k_encode(const float* __restrict__ x, const float* __restrict__ t,
                         const float* __restrict__ Wenc, float* __restrict__ nodes) {
  int idx = blockIdx.x * blockDim.x + threadIdx.x;
  if (idx >= B_ * N_ * DIM_) return;
  int c = idx % DIM_;
  int row = idx / DIM_;
  const float* xr = x + row * INNF_;
  float acc = xr[0] * Wenc[0 * DIM_ + c] + xr[1] * Wenc[1 * DIM_ + c] +
              xr[2] * Wenc[2 * DIM_ + c] + t[row] * Wenc[3 * DIM_ + c];
  nodes[idx] = acc;
}

// --------------------------------------------- weight convert (transpose->f16)
__global__ void k_cvt_wall(const float* __restrict__ Wq, const float* __restrict__ Wkv,
                           _Float16* __restrict__ WallT) {
  int idx = blockIdx.x * blockDim.x + threadIdx.x;   // 768*256
  if (idx >= 768 * 256) return;
  int k = idx % 256;
  int n = idx / 256;
  float v = (n < 256) ? Wq[k * 256 + n] : Wkv[k * 512 + (n - 256)];
  WallT[idx] = (_Float16)v;
}

__global__ void k_cvt_wo(const float* __restrict__ Wo, _Float16* __restrict__ WoT) {
  int idx = blockIdx.x * blockDim.x + threadIdx.x;   // 256*256
  if (idx >= 256 * 256) return;
  int k = idx % 256;
  int n = idx / 256;
  WoT[idx] = (_Float16)Wo[k * 256 + n];
}

// ---------------------------------------------------------------- layernorm
__global__ void k_ln(const float* __restrict__ nodes, const float* __restrict__ g,
                     const float* __restrict__ bb, _Float16* __restrict__ h16) {
  __shared__ float red[DIM_];
  int row = blockIdx.x, tid = threadIdx.x;
  float v = nodes[row * DIM_ + tid];
  red[tid] = v;
  __syncthreads();
  for (int s = 128; s > 0; s >>= 1) { if (tid < s) red[tid] += red[tid + s]; __syncthreads(); }
  float mean = red[0] * (1.0f / DIM_);
  __syncthreads();
  float d = v - mean;
  red[tid] = d * d;
  __syncthreads();
  for (int s = 128; s > 0; s >>= 1) { if (tid < s) red[tid] += red[tid + s]; __syncthreads(); }
  float r = rsqrtf(red[0] * (1.0f / DIM_) + 1e-5f);
  h16[row * DIM_ + tid] = (_Float16)(d * r * g[tid] + bb[tid]);
}

// -------------------------------------- fused QKV GEMM + bias + rotary (WMMA)
// wave tile: 16 rows x 32 cols (one output head). cols 0-255=q, 256-511=k, 512-767=v
__global__ void __launch_bounds__(128) k_qkv(
    const _Float16* __restrict__ h16, const _Float16* __restrict__ WallT,
    const float* __restrict__ bq, const float* __restrict__ bkv,
    _Float16* __restrict__ q16, _Float16* __restrict__ k16,
    _Float16* __restrict__ vT16) {
  int wid = blockIdx.x * (blockDim.x >> 5) + (threadIdx.x >> 5);  // < 64*24
  int lane = threadIdx.x & 31, lo = lane & 15, hi = lane >> 4;
  int mt = wid / 24, ct = wid % 24;

  v8f c0 = {}, c1 = {};
  const _Float16* arow  = h16 + (mt * 16 + lo) * DIM_;
  const _Float16* brow0 = WallT + (ct * 32 + lo) * DIM_;
  const _Float16* brow1 = brow0 + 16 * DIM_;
  for (int kk = 0; kk < DIM_; kk += 32) {
    v16h a, b0, b1;
#pragma unroll
    for (int e = 0; e < 8; ++e) {
      a[e]     = arow[kk + 16 * hi + e];
      a[8 + e] = arow[kk + 16 * hi + 8 + e];
    }
#pragma unroll
    for (int e = 0; e < 16; ++e) {
      b0[e] = brow0[kk + 16 * hi + e];
      b1[e] = brow1[kk + 16 * hi + e];
    }
    c0 = wmma16(a, b0, c0);
    c1 = wmma16(a, b1, c1);
  }
  int cbase = ct * 32;
  float bias0, bias1;
  if (ct < 8) { bias0 = bq[cbase + lo];        bias1 = bq[cbase + lo + 16]; }
  else        { bias0 = bkv[cbase - 256 + lo]; bias1 = bkv[cbase - 256 + lo + 16]; }
#pragma unroll
  for (int e = 0; e < 8; ++e) { c0[e] += bias0; c1[e] += bias1; }

  if (ct < 16) {  // q or k: rotary then store [b][h][i][d]
    float invf = __powf(10000.0f, -(float)lo * (1.0f / 16.0f));
#pragma unroll
    for (int e = 0; e < 8; ++e) {
      int rowg = mt * 16 + e + 8 * hi;
      float pos = (float)(rowg & (N_ - 1));
      float ang = pos * invf;
      float cs = __cosf(ang), sn = __sinf(ang);
      float v0 = c0[e] * cs - c1[e] * sn;
      float v1 = c1[e] * cs + c0[e] * sn;
      c0[e] = v0; c1[e] = v1;
    }
    int head = ct & 7;
    _Float16* dst = (ct < 8) ? q16 : k16;
#pragma unroll
    for (int e = 0; e < 8; ++e) {
      int rowg = mt * 16 + e + 8 * hi;
      int b = rowg >> 9, i = rowg & (N_ - 1);
      _Float16* p = dst + (((b * HEADS_ + head) * N_ + i) << 5);
      p[lo]      = (_Float16)c0[e];
      p[lo + 16] = (_Float16)c1[e];
    }
  } else {        // v: store transposed [b][h][d][j]
    int head = ct - 16;
#pragma unroll
    for (int e = 0; e < 8; ++e) {
      int rowg = mt * 16 + e + 8 * hi;
      int b = rowg >> 9, i = rowg & (N_ - 1);
      _Float16* p = vT16 + (size_t)((b * HEADS_ + head) << 5) * N_;
      p[lo * N_ + i]        = (_Float16)c0[e];
      p[(lo + 16) * N_ + i] = (_Float16)c1[e];
    }
  }
}

// ------------------------------------------- qw = q.We_h , qb = q.be_h
__global__ void k_qwqb(const _Float16* __restrict__ q16, const float* __restrict__ We,
                       const float* __restrict__ be, float* __restrict__ qw,
                       float* __restrict__ qb) {
  int idx = blockIdx.x * blockDim.x + threadIdx.x;   // (b*H+h)*N + i
  if (idx >= B_ * HEADS_ * N_) return;
  int h = (idx / N_) % HEADS_;
  const _Float16* q = q16 + ((size_t)idx << 5);
  float sw = 0.0f, sb = 0.0f;
#pragma unroll
  for (int d = 0; d < DHEAD_; ++d) {
    float qv = (float)q[d];
    sw += qv * We[h * DHEAD_ + d];
    sb += qv * be[h * DHEAD_ + d];
  }
  qw[idx] = sw;
  qb[idx] = sb;
}

// --------------------- flash attention with scalar dist bias + rank-1 e term
__global__ void __launch_bounds__(128) k_attn(
    const _Float16* __restrict__ q16, const _Float16* __restrict__ k16,
    const _Float16* __restrict__ vT16, const float* __restrict__ dist,
    const float* __restrict__ qw, const float* __restrict__ qb,
    const float* __restrict__ We, const float* __restrict__ be,
    _Float16* __restrict__ attout) {
  __shared__ _Float16 pbuf[4][16 * 32];
  int wv = threadIdx.x >> 5;
  int wid = blockIdx.x * 4 + wv;                    // < B*H*(N/16) = 512
  int lane = threadIdx.x & 31, lo = lane & 15, hi = lane >> 4;
  int it = wid & 31;
  int h  = (wid >> 5) & 7;
  int b  = wid >> 8;
  int bh = b * HEADS_ + h;
  int i0 = it * 16;
  _Float16* pb = pbuf[wv];

  const _Float16* qrow = q16 + ((size_t)(bh * N_ + i0 + lo) << 5);
  v16h aq;
#pragma unroll
  for (int e = 0; e < 8; ++e) {
    aq[e]     = qrow[16 * hi + e];
    aq[8 + e] = qrow[16 * hi + 8 + e];
  }

  float m[8], lsum[8], adl[8], qwr[8], qbr[8];
  v8f o0 = {}, o1 = {};
#pragma unroll
  for (int e = 0; e < 8; ++e) {
    m[e] = -1e30f; lsum[e] = 0.0f; adl[e] = 0.0f;
    int i = i0 + e + 8 * hi;
    qwr[e] = qw[bh * N_ + i];
    qbr[e] = qb[bh * N_ + i];
  }

  for (int j0 = 0; j0 < N_; j0 += 32) {
    const _Float16* k0p = k16 + ((size_t)(bh * N_ + j0 + lo) << 5) + 16 * hi;
    const _Float16* k1p = k0p + (16 << 5);
    v16h bk0, bk1;
#pragma unroll
    for (int e = 0; e < 16; ++e) { bk0[e] = k0p[e]; bk1[e] = k1p[e]; }
    v8f z = {};
    v8f s0 = wmma16(aq, bk0, z);
    v8f s1 = wmma16(aq, bk1, z);

    float alpha[8];
#pragma unroll
    for (int e = 0; e < 8; ++e) {
      int i = i0 + e + 8 * hi;
      float dv0 = dist[(size_t)(b * N_ + i) * N_ + j0 + lo];
      float dv1 = dist[(size_t)(b * N_ + i) * N_ + j0 + 16 + lo];
      float v0 = (s0[e] + dv0 * qwr[e] + qbr[e]) * SCALE_;
      float v1 = (s1[e] + dv1 * qwr[e] + qbr[e]) * SCALE_;
      float rm = fmaxf(v0, v1);
      rm = fmaxf(rm, __shfl_xor(rm, 1, 32));
      rm = fmaxf(rm, __shfl_xor(rm, 2, 32));
      rm = fmaxf(rm, __shfl_xor(rm, 4, 32));
      rm = fmaxf(rm, __shfl_xor(rm, 8, 32));
      float mn = fmaxf(m[e], rm);
      float al = __expf(m[e] - mn);
      m[e] = mn; alpha[e] = al;
      float p0 = __expf(v0 - mn);
      float p1 = __expf(v1 - mn);
      float rs = p0 + p1;
      rs += __shfl_xor(rs, 1, 32);
      rs += __shfl_xor(rs, 2, 32);
      rs += __shfl_xor(rs, 4, 32);
      rs += __shfl_xor(rs, 8, 32);
      lsum[e] = lsum[e] * al + rs;
      adl[e]  = adl[e]  * al + p0 * dv0 + p1 * dv1;
      pb[(e + 8 * hi) * 32 + lo]      = (_Float16)p0;
      pb[(e + 8 * hi) * 32 + lo + 16] = (_Float16)p1;
    }
    // reload p tile in A-operand layout (same-wave LDS round trip)
    v16h ap;
    const _Float16* prow = pb + lo * 32;
#pragma unroll
    for (int e = 0; e < 8; ++e) {
      ap[e]     = prow[16 * hi + e];
      ap[8 + e] = prow[16 * hi + 8 + e];
    }
    const _Float16* v0p = vT16 + (size_t)(bh * 32 + lo) * N_ + j0 + 16 * hi;
    const _Float16* v1p = vT16 + (size_t)(bh * 32 + lo + 16) * N_ + j0 + 16 * hi;
    v16h bv0, bv1;
#pragma unroll
    for (int e = 0; e < 16; ++e) { bv0[e] = v0p[e]; bv1[e] = v1p[e]; }
#pragma unroll
    for (int e = 0; e < 8; ++e) { o0[e] *= alpha[e]; o1[e] *= alpha[e]; }
    o0 = wmma16(ap, bv0, o0);
    o1 = wmma16(ap, bv1, o1);
  }

  float we0 = We[h * DHEAD_ + lo],  we1 = We[h * DHEAD_ + lo + 16];
  float be0 = be[h * DHEAD_ + lo],  be1 = be[h * DHEAD_ + lo + 16];
#pragma unroll
  for (int e = 0; e < 8; ++e) {
    float ad = adl[e];
    ad += __shfl_xor(ad, 1, 32);
    ad += __shfl_xor(ad, 2, 32);
    ad += __shfl_xor(ad, 4, 32);
    ad += __shfl_xor(ad, 8, 32);
    float linv = 1.0f / lsum[e];
    int i = i0 + e + 8 * hi;
    _Float16* op = attout + (size_t)(b * N_ + i) * INNER_ + h * DHEAD_;
    op[lo]      = (_Float16)(o0[e] * linv + ad * linv * we0 + be0);
    op[lo + 16] = (_Float16)(o1[e] * linv + ad * linv * we1 + be1);
  }
}

// ------------------------------------------------ out projection (WMMA GEMM)
__global__ void __launch_bounds__(128) k_oproj(
    const _Float16* __restrict__ a16, const _Float16* __restrict__ WoT,
    const float* __restrict__ bo, float* __restrict__ outb) {
  int wid = blockIdx.x * (blockDim.x >> 5) + (threadIdx.x >> 5);  // < 64*8
  int lane = threadIdx.x & 31, lo = lane & 15, hi = lane >> 4;
  int mt = wid >> 3, ct = wid & 7;
  v8f c0 = {}, c1 = {};
  const _Float16* arow  = a16 + (mt * 16 + lo) * INNER_;
  const _Float16* brow0 = WoT + (ct * 32 + lo) * INNER_;
  const _Float16* brow1 = brow0 + 16 * INNER_;
  for (int kk = 0; kk < INNER_; kk += 32) {
    v16h a, b0, b1;
#pragma unroll
    for (int e = 0; e < 8; ++e) {
      a[e]     = arow[kk + 16 * hi + e];
      a[8 + e] = arow[kk + 16 * hi + 8 + e];
    }
#pragma unroll
    for (int e = 0; e < 16; ++e) {
      b0[e] = brow0[kk + 16 * hi + e];
      b1[e] = brow1[kk + 16 * hi + e];
    }
    c0 = wmma16(a, b0, c0);
    c1 = wmma16(a, b1, c1);
  }
  float bias0 = bo[ct * 32 + lo], bias1 = bo[ct * 32 + lo + 16];
#pragma unroll
  for (int e = 0; e < 8; ++e) {
    int rowg = mt * 16 + e + 8 * hi;
    outb[(size_t)rowg * DIM_ + ct * 32 + lo]      = c0[e] + bias0;
    outb[(size_t)rowg * DIM_ + ct * 32 + lo + 16] = c1[e] + bias1;
  }
}

// ------------------------------------------------------- gated residual
__global__ void k_gate(const float* __restrict__ outb, float* __restrict__ nodes,
                       const float* __restrict__ Wg) {
  __shared__ float red[DIM_];
  int row = blockIdx.x, c = threadIdx.x;
  float ov = outb[row * DIM_ + c];
  float nv = nodes[row * DIM_ + c];
  red[c] = ov * (Wg[c] + Wg[512 + c]) + nv * (Wg[256 + c] - Wg[512 + c]);
  __syncthreads();
  for (int s = 128; s > 0; s >>= 1) { if (c < s) red[c] += red[c + s]; __syncthreads(); }
  float g = 1.0f / (1.0f + __expf(-red[0]));
  nodes[row * DIM_ + c] = ov * g + nv * (1.0f - g);
}

// ------------------------------------------------------------- decode
__global__ void k_decode(const float* __restrict__ nodes, const float* __restrict__ Wdec,
                         float* __restrict__ out) {
  int idx = blockIdx.x * blockDim.x + threadIdx.x;
  if (idx >= B_ * N_ * INNF_) return;
  int c = idx % INNF_;
  int row = idx / INNF_;
  float acc = 0.0f;
  for (int k = 0; k < DIM_; ++k) acc += nodes[row * DIM_ + k] * Wdec[k * INNF_ + c];
  out[idx] = acc;
}

// ================================================================= launch
extern "C" void kernel_launch(void* const* d_in, const int* in_sizes, int n_in,
                              void* d_out, int out_size, void* d_ws, size_t ws_size,
                              hipStream_t stream) {
  (void)in_sizes; (void)n_in; (void)out_size; (void)ws_size;
  const float* x    = (const float*)d_in[0];
  const float* t    = (const float*)d_in[1];
  const float* Wenc = (const float*)d_in[2];
  const float* Wdec = (const float*)d_in[3];
  const float* ln_g = (const float*)d_in[4];
  const float* ln_b = (const float*)d_in[5];
  const float* Wq   = (const float*)d_in[6];
  const float* bq   = (const float*)d_in[7];
  const float* Wkv  = (const float*)d_in[8];
  const float* bkv  = (const float*)d_in[9];
  const float* We   = (const float*)d_in[10];
  const float* be   = (const float*)d_in[11];
  const float* Wo   = (const float*)d_in[12];
  const float* bo   = (const float*)d_in[13];
  const float* Wg   = (const float*)d_in[14];

  char* ws = (char*)d_ws;
  float*    dist     = (float*)   (ws + 0);          // 2 MB
  float*    nodes    = (float*)   (ws + 2097152);    // 1 MB
  _Float16* h16      = (_Float16*)(ws + 3145728);    // 0.5 MB
  _Float16* q16      = (_Float16*)(ws + 3670016);    // 0.5 MB
  _Float16* k16      = (_Float16*)(ws + 4194304);    // 0.5 MB
  _Float16* vT16     = (_Float16*)(ws + 4718592);    // 0.5 MB
  float*    qwb      = (float*)   (ws + 5242880);    // 32 KB
  float*    qbb      = (float*)   (ws + 5275648);    // 32 KB
  _Float16* attout16 = (_Float16*)(ws + 5308416);    // 0.5 MB
  float*    outb     = (float*)   (ws + 5832704);    // 1 MB
  _Float16* WallT    = (_Float16*)(ws + 6881280);    // 384 KB
  _Float16* WoT      = (_Float16*)(ws + 7274496);    // 128 KB

  k_dist  <<<2048, 256, 0, stream>>>(x, dist);
  k_encode<<<1024, 256, 0, stream>>>(x, t, Wenc, nodes);

  for (int l = 0; l < DEPTH_; ++l) {
    k_cvt_wall<<<768, 256, 0, stream>>>(Wq + l * 256 * 256, Wkv + l * 256 * 512, WallT);
    k_cvt_wo  <<<256, 256, 0, stream>>>(Wo + l * 256 * 256, WoT);
    k_ln      <<<1024, 256, 0, stream>>>(nodes, ln_g + l * 256, ln_b + l * 256, h16);
    k_qkv     <<<384, 128, 0, stream>>>(h16, WallT, bq + l * 256, bkv + l * 512,
                                        q16, k16, vT16);
    k_qwqb    <<<32, 256, 0, stream>>>(q16, We + l * 256, be + l * 256, qwb, qbb);
    k_attn    <<<128, 128, 0, stream>>>(q16, k16, vT16, dist, qwb, qbb,
                                        We + l * 256, be + l * 256, attout16);
    k_oproj   <<<128, 128, 0, stream>>>(attout16, WoT, bo + l * 256, outb);
    k_gate    <<<1024, 256, 0, stream>>>(outb, nodes, Wg + l * 768);
  }
  k_decode<<<12, 256, 0, stream>>>(nodes, Wdec, (float*)d_out);
}
